// GatedAttention_64819646431749
// MI455X (gfx1250) — matrix operand order repaired and checked by reading, not compiled, not run on hardware
//
#include <hip/hip_runtime.h>
#include <cstdint>
#include <cstddef>

typedef __bf16 bf16_t;
typedef __attribute__((ext_vector_type(16))) __bf16 v16bf;
typedef __attribute__((ext_vector_type(8)))  float  v8f;

#define B_    2
#define L_    2048
#define HID_  2048
#define NH_   16
#define HD_   128
#define SCALE_ 0.08838834764831845f
#define EPS_   1e-5f
#define NEG_BIG -3.0e38f

// ---------------------------------------------------------------------------
// WMMA fragment loaders (contiguous k order, same for A and B)
// A 16x32 bf16: lane m = lane&15; half = lane>>4;
//   element s holds K = half*16 + s (matches load_bfrag)
// Per lane: 16 contiguous bf16 -> 2x b128 loads.
__device__ inline v16bf load_afrag(const bf16_t* A, int lda, int k0, int lane) {
    int half = lane >> 4;
    int m    = lane & 15;
    const bf16_t* p = A + (size_t)m * lda + k0;
    v16bf a;
#pragma unroll
    for (int j = 0; j < 8; ++j) {
        int kk = (half << 4) + (j << 1);
        a[2 * j]     = p[kk];
        a[2 * j + 1] = p[kk + 1];
    }
    return a;
}

// B 32x16 bf16 (Wm is row-major [N][K], i.e. column n of B = row n of Wm):
//   lane n = lane&15; lanes 0-15 hold K=k0..k0+15, lanes 16-31 hold K=k0+16..k0+31
// Per lane: 16 contiguous bf16 -> 2x b128 loads.
__device__ inline v16bf load_bfrag(const bf16_t* Wm, int ldb, int k0, int lane) {
    int half = lane >> 4;
    int n    = lane & 15;
    const bf16_t* p = Wm + (size_t)n * ldb + k0 + (half << 4);
    v16bf b;
#pragma unroll
    for (int j = 0; j < 16; ++j) b[j] = p[j];
    return b;
}

// ---------------------------------------------------------------------------
// fp32 -> bf16 convert
__global__ void cvt_k(const float* __restrict__ in, bf16_t* __restrict__ out, int n) {
    int i = blockIdx.x * blockDim.x + threadIdx.x;
    if (i < n) out[i] = (bf16_t)in[i];
}

// ---------------------------------------------------------------------------
// Y[M,N] = X[M,K] * W[N,K]^T, bf16 inputs, fp32 accumulate.
// Wave computes a 16x64 tile (4 accumulators). 8 waves / block.
// HEADOUT=false: store fp32 row-major [M,N].  HEADOUT=true: store bf16 in
// per-head layout [B,NH,L,HD] (M = b*L+l, N = h*HD+d).
template <bool HEADOUT>
__global__ __launch_bounds__(256, 1)
void gemm_bf16_k(const bf16_t* __restrict__ A, const bf16_t* __restrict__ W,
                 float* __restrict__ outF, bf16_t* __restrict__ outH,
                 int M, int N, int K) {
    int lane = threadIdx.x & 31;
    int wid  = blockIdx.x * (blockDim.x >> 5) + (threadIdx.x >> 5);
    int ntn  = N >> 6;
    int tm   = (wid / ntn) << 4;
    int tn   = (wid % ntn) << 6;
    if (tm >= M) return;

    v8f acc[4] = {};
    const bf16_t* Arow = A + (size_t)tm * K;
    for (int k0 = 0; k0 < K; k0 += 32) {
        v16bf a = load_afrag(Arow, K, k0, lane);
#pragma unroll
        for (int j = 0; j < 4; ++j) {
            v16bf b = load_bfrag(W + (size_t)(tn + j * 16) * K, K, k0, lane);
            acc[j] = __builtin_amdgcn_wmma_f32_16x16x32_bf16(
                false, a, false, b, (short)0, acc[j], false, false);
        }
    }

    int half = lane >> 4, l15 = lane & 15;
#pragma unroll
    for (int j = 0; j < 4; ++j) {
#pragma unroll
        for (int r = 0; r < 8; ++r) {
            int m = tm + r + (half << 3);
            int n = tn + j * 16 + l15;
            float v = acc[j][r];
            if (HEADOUT) {
                int b = m >> 11;            // L_ == 2048
                int l = m & (L_ - 1);
                int h = n >> 7;             // HD_ == 128
                int d = n & (HD_ - 1);
                outH[(((size_t)(b * NH_ + h) * L_ + l) << 7) + d] = (bf16_t)v;
            } else {
                outF[(size_t)m * N + n] = v;
            }
        }
    }
}

// ---------------------------------------------------------------------------
// RoPE: read fp32 stage [B,L,HID] (n = h*HD+d), write bf16 [B,NH,L,HD]
__global__ void rope_k(const float* __restrict__ stage, bf16_t* __restrict__ out) {
    int idx = blockIdx.x * blockDim.x + threadIdx.x;   // B*NH*L*HD threads
    int d = idx & 127;
    int l = (idx >> 7) & (L_ - 1);
    int h = (idx >> 18) & (NH_ - 1);
    int b = idx >> 22;
    const float* row = stage + ((size_t)(b * L_ + l)) * HID_ + h * HD_;
    float xv  = row[d];
    float rot = (d < 64) ? -row[d + 64] : row[d - 64];
    int   fi  = d & 63;
    // inv_freq = 10000^(-2*fi/128) = exp(-2*fi * ln(10000)/128)
    float inv = expf(-(float)(2 * fi) * (9.210340371976184f / 128.0f));
    float ang = (float)l * inv;
    out[idx] = (bf16_t)(xv * cosf(ang) + rot * sinf(ang));
}

// V transpose: fp32 stage [B,L,HID] -> bf16 [B,NH,HD,L], LDS-tiled 32x32 so
// both global reads (along d) and global writes (along l) are coalesced.
__global__ void vtrans_k(const float* __restrict__ stage, bf16_t* __restrict__ vt) {
    __shared__ float tile[32][33];
    int bh = blockIdx.z;                       // b*NH + h
    int l0 = blockIdx.x << 5;
    int d0 = blockIdx.y << 5;
    int tx = threadIdx.x;                      // 0..31
    int ty = threadIdx.y;                      // 0..7
    int h = bh & (NH_ - 1);
    int b = bh >> 4;
    const float* src = stage + ((size_t)b * L_) * HID_ + h * HD_;
#pragma unroll
    for (int i = 0; i < 32; i += 8) {
        int l = l0 + ty + i;
        tile[ty + i][tx] = src[(size_t)l * HID_ + d0 + tx];
    }
    __syncthreads();
    bf16_t* dst = vt + ((size_t)bh * HD_) * L_;
#pragma unroll
    for (int i = 0; i < 32; i += 8) {
        int d = d0 + ty + i;
        dst[(size_t)d * L_ + l0 + tx] = (bf16_t)tile[tx][ty + i];
    }
}

// ---------------------------------------------------------------------------
// Flash attention + RMSNorm + SiLU gate. One wave per 16-query tile.
// qm,km: [B,NH,L,HD] bf16 (rope'd). vm: [B,NH,HD,L] bf16. gm: [B,NH,L,HD] bf16.
// outb: [B,L,NH*HD] bf16 (input to final projection).
__global__ __launch_bounds__(256, 1)
void attn_k(const bf16_t* __restrict__ qm, const bf16_t* __restrict__ km,
            const bf16_t* __restrict__ vm, const bf16_t* __restrict__ gm,
            const float* __restrict__ nw, bf16_t* __restrict__ outb) {
    __shared__ bf16_t ldsP[8][16 * 32];

    int lane  = threadIdx.x & 31;
    int wslot = threadIdx.x >> 5;
    int wid   = blockIdx.x * (blockDim.x >> 5) + wslot;
    int qt = wid & (L_ / 16 - 1);          // 128 q-tiles per head
    int h  = (wid >> 7) & (NH_ - 1);
    int b  = wid >> 11;
    int qbase = qt << 4;
    int half = lane >> 4, l15 = lane & 15;

    const bf16_t* qh = qm + ((size_t)(b * NH_ + h) * L_ + qbase) * HD_;
    const bf16_t* kh = km + ((size_t)(b * NH_ + h) * L_) * HD_;
    const bf16_t* vh = vm + ((size_t)(b * NH_ + h) * HD_) * L_;
    const bf16_t* gh = gm + ((size_t)(b * NH_ + h) * L_) * HD_;

    v16bf qf[4];
#pragma unroll
    for (int t = 0; t < 4; ++t) qf[t] = load_afrag(qh, HD_, t * 32, lane);

    v8f acc[8] = {};
    float mrow[8], lrow[8];
#pragma unroll
    for (int r = 0; r < 8; ++r) { mrow[r] = NEG_BIG; lrow[r] = 0.f; }

    int kend = qbase + 16;
    for (int kb0 = 0; kb0 < kend; kb0 += 32) {
        v8f S0 = {}, S1 = {};
#pragma unroll
        for (int t = 0; t < 4; ++t) {
            v16bf bk0 = load_bfrag(kh + (size_t)kb0 * HD_,        HD_, t * 32, lane);
            v16bf bk1 = load_bfrag(kh + (size_t)(kb0 + 16) * HD_, HD_, t * 32, lane);
            S0 = __builtin_amdgcn_wmma_f32_16x16x32_bf16(false, qf[t], false, bk0, (short)0, S0, false, false);
            S1 = __builtin_amdgcn_wmma_f32_16x16x32_bf16(false, qf[t], false, bk1, (short)0, S1, false, false);
        }
        // scale + causal mask + online softmax (rows live across 16-lane half-groups)
#pragma unroll
        for (int r = 0; r < 8; ++r) {
            int qrow = qbase + r + (half << 3);
            int key0 = kb0 + l15;
            int key1 = kb0 + 16 + l15;
            float s0 = (key0 <= qrow) ? S0[r] * SCALE_ : NEG_BIG;
            float s1 = (key1 <= qrow) ? S1[r] * SCALE_ : NEG_BIG;
            float rm = fmaxf(s0, s1);
#pragma unroll
            for (int off = 8; off; off >>= 1) rm = fmaxf(rm, __shfl_xor(rm, off, 16));
            float mn    = fmaxf(mrow[r], rm);
            float alpha = __expf(mrow[r] - mn);
            float p0 = __expf(s0 - mn);
            float p1 = __expf(s1 - mn);
            float rs = p0 + p1;
#pragma unroll
            for (int off = 8; off; off >>= 1) rs += __shfl_xor(rs, off, 16);
            lrow[r] = lrow[r] * alpha + rs;
            mrow[r] = mn;
#pragma unroll
            for (int f = 0; f < 8; ++f) acc[f][r] *= alpha;
            int row = r + (half << 3);
            ldsP[wslot][row * 32 + l15]      = (bf16_t)p0;
            ldsP[wslot][row * 32 + 16 + l15] = (bf16_t)p1;
        }
        asm volatile("s_wait_dscnt 0" ::: "memory");
        v16bf aP = load_afrag(&ldsP[wslot][0], 32, 0, lane);
#pragma unroll
        for (int f = 0; f < 8; ++f) {
            v16bf bv = load_bfrag(vh + (size_t)(f * 16) * L_, L_, kb0, lane);
            acc[f] = __builtin_amdgcn_wmma_f32_16x16x32_bf16(false, aP, false, bv, (short)0, acc[f], false, false);
        }
    }

    // Epilogue: 1/l, RMSNorm over HD, norm_w, SiLU(g) gate, store bf16.
    float nwv[8];
#pragma unroll
    for (int f = 0; f < 8; ++f) nwv[f] = nw[f * 16 + l15];

#pragma unroll
    for (int r = 0; r < 8; ++r) {
        float inv_l = 1.0f / lrow[r];
        float ov[8];
        float ss = 0.f;
#pragma unroll
        for (int f = 0; f < 8; ++f) { ov[f] = acc[f][r] * inv_l; ss += ov[f] * ov[f]; }
#pragma unroll
        for (int off = 8; off; off >>= 1) ss += __shfl_xor(ss, off, 16);
        float rms = rsqrtf(ss * (1.0f / HD_) + EPS_);
        int row = qbase + r + (half << 3);
#pragma unroll
        for (int f = 0; f < 8; ++f) {
            float g   = (float)gh[(size_t)row * HD_ + f * 16 + l15];
            float sil = g / (1.0f + __expf(-g));
            float val = ov[f] * rms * nwv[f] * sil;
            outb[(((size_t)(b * L_ + row)) << 11) + (h << 7) + f * 16 + l15] = (bf16_t)val;
        }
    }
}

// ---------------------------------------------------------------------------
extern "C" void kernel_launch(void* const* d_in, const int* in_sizes, int n_in,
                              void* d_out, int out_size, void* d_ws, size_t ws_size,
                              hipStream_t stream) {
    (void)in_sizes; (void)n_in; (void)out_size; (void)ws_size;
    const float* x  = (const float*)d_in[0];
    const float* wq = (const float*)d_in[1];
    const float* wk = (const float*)d_in[2];
    const float* wv = (const float*)d_in[3];
    const float* wg = (const float*)d_in[4];
    const float* wo = (const float*)d_in[5];
    const float* nw = (const float*)d_in[6];
    float* out = (float*)d_out;

    const int M = B_ * L_;       // 4096
    const int N = HID_;          // 2048
    const int K = HID_;          // 2048
    const int nX = M * K;        // 8,388,608  (== B*NH*L*HD)
    const int nW = N * K;        // 4,194,304

    char* p = (char*)d_ws;
    bf16_t* xb    = (bf16_t*)(p);                      // 16,777,216 B
    bf16_t* wbuf  = (bf16_t*)(p +  16777216);          //  8,388,608 B (reused per weight)
    float*  stage = (float*) (p +  25165824);          // 33,554,432 B (reused)
    bf16_t* qbuf  = (bf16_t*)(p +  58720256);          // 16,777,216 B
    bf16_t* kbuf  = (bf16_t*)(p +  75497472);          // 16,777,216 B
    bf16_t* vtb   = (bf16_t*)(p +  92274688);          // 16,777,216 B
    bf16_t* gbuf  = (bf16_t*)(p + 109051904);          // 16,777,216 B
    bf16_t* outb  = (bf16_t*)(p + 125829120);          // 16,777,216 B

    const int gemm_blocks = (M / 16) * (N / 64) / 8;   // 1024 (8 waves/block)
    dim3 tr_grid(L_ / 32, HD_ / 32, B_ * NH_);
    dim3 tr_block(32, 8);

    cvt_k<<<nX / 256, 256, 0, stream>>>(x, xb, nX);

    // Q projection + RoPE
    cvt_k<<<nW / 256, 256, 0, stream>>>(wq, wbuf, nW);
    gemm_bf16_k<false><<<gemm_blocks, 256, 0, stream>>>(xb, wbuf, stage, nullptr, M, N, K);
    rope_k<<<nX / 256, 256, 0, stream>>>(stage, qbuf);

    // K projection + RoPE
    cvt_k<<<nW / 256, 256, 0, stream>>>(wk, wbuf, nW);
    gemm_bf16_k<false><<<gemm_blocks, 256, 0, stream>>>(xb, wbuf, stage, nullptr, M, N, K);
    rope_k<<<nX / 256, 256, 0, stream>>>(stage, kbuf);

    // V projection + transpose to [B,NH,HD,L]
    cvt_k<<<nW / 256, 256, 0, stream>>>(wv, wbuf, nW);
    gemm_bf16_k<false><<<gemm_blocks, 256, 0, stream>>>(xb, wbuf, stage, nullptr, M, N, K);
    vtrans_k<<<tr_grid, tr_block, 0, stream>>>(stage, vtb);

    // Gate projection, direct bf16 head layout
    cvt_k<<<nW / 256, 256, 0, stream>>>(wg, wbuf, nW);
    gemm_bf16_k<true><<<gemm_blocks, 256, 0, stream>>>(xb, wbuf, nullptr, gbuf, M, N, K);

    // Flash attention + RMSNorm + SiLU gate
    const int attn_blocks = (B_ * NH_ * (L_ / 16)) / 8;  // 512
    attn_k<<<attn_blocks, 256, 0, stream>>>(qbuf, kbuf, vtb, gbuf, nw, outb);

    // Output projection -> fp32 d_out
    cvt_k<<<nW / 256, 256, 0, stream>>>(wo, wbuf, nW);
    gemm_bf16_k<false><<<gemm_blocks, 256, 0, stream>>>(outb, wbuf, out, nullptr, M, N, K);
}
